// MoE_CondNP_21612275433874
// MI455X (gfx1250) — matrix operand order repaired and checked by reading, not compiled, not run on hardware
//
#include <hip/hip_runtime.h>
#include <math.h>

// ---------------------------------------------------------------------------
// MoE Conditional Neural Process forward, MI455X (gfx1250, wave32, WMMA).
// Heavy GEMMs: V_WMMA_F32_16X16X32_BF16, tiles staged with
// GLOBAL_LOAD_ASYNC_TO_LDS_B128 (ASYNCcnt) from pre-transposed bf16 weights.
// ---------------------------------------------------------------------------

typedef __bf16 bf16;
typedef __attribute__((ext_vector_type(16))) __bf16 v16bf;
typedef __attribute__((ext_vector_type(8)))  float  v8f;

// Problem constants
#define Bsz   8
#define Tsz   1024
#define Ksz   8
#define Lsz   128
#define Hsz   512
#define HGsz  256
#define DHsz  512
#define NPTS  512           // NC == NT
#define ROWS  (Bsz*Tsz*Ksz) // 65536 gate/decoder rows

// Output layout (floats), reference return order
#define OUT_MU_C   0
#define OUT_LV_C   8192
#define OUT_MU_T   16384
#define OUT_LV_T   24576
#define OUT_YMEAN  32768
#define OUT_YSTD   229376
#define OUT_APOST  425984
#define OUT_APRIOR 491520

// ---------------------------------------------------------------------------
// Generic fused GEMM:  C = relu(A @ W + bias)   (bf16 in, bf16 out, f32 acc)
//   A : [G>>aShift][M][Kd]   row-major, Kd multiple of 32
//   WT: [(wMask+1)][N][Kd]   TRANSPOSED weights (row n holds column n of W)
//   grid = (tilesM*tilesN, G), 256 threads (8 waves, 2x4 wave grid)
//   epi==1: encoder mean-pool epilogue -> deterministic partial sums in pp
// ---------------------------------------------------------------------------
__global__ __launch_bounds__(256)
void gemm_bf16(const bf16* __restrict__ A, const bf16* __restrict__ WT,
               const float* __restrict__ bias, bf16* __restrict__ C,
               float* __restrict__ pp,
               int M, int Kd, int N, int tilesM, int aShift, int wMask, int epi)
{
    __shared__ bf16 sA[2][128][40];   // A tile  [m][k], k-pad 40 (80B rows, 16B aligned)
    __shared__ bf16 sW[2][128][40];   // W^T tile [n][k], same layout

    const int g     = blockIdx.y;
    const int tileM = blockIdx.x % tilesM;
    const int tileN = blockIdx.x / tilesM;

    const bf16*  Ag = A  + (size_t)(g >> aShift) * M * Kd;
    const bf16*  Wg = WT + (size_t)(g & wMask) * N * Kd;
    const float* bg = bias + (size_t)(g & wMask) * N;
    bf16*        Cg = C + (size_t)g * M * N;

    const int tid  = threadIdx.x;
    const int lane = tid & 31;
    const int wave = tid >> 5;
    const int wm   = wave >> 2;   // 0..1  -> 64-row slab
    const int wn   = wave & 3;    // 0..3  -> 32-col slab
    const int c16  = lane & 15;
    const int half = lane >> 4;

    // Async stage one K-step of both tiles: 128 rows x 32 bf16 each, 16B chunks.
    // GLOBAL_LOAD_ASYNC_TO_LDS_B128: memory -> LDS, no VGPR data, ASYNCcnt.
    auto stage = [&](int buf, int kk) {
        const bf16* aSrc = Ag + (size_t)tileM * 128 * Kd + (size_t)kk * 32;
        const bf16* wSrc = Wg + (size_t)tileN * 128 * Kd + (size_t)kk * 32;
#pragma unroll
        for (int it = 0; it < 2; ++it) {
            int chunk = tid + it * 256;           // 512 chunks of 8 bf16 per tile
            int row   = chunk >> 2;
            int kc    = (chunk & 3) * 8;
            unsigned la = (unsigned)(size_t)&sA[buf][row][kc];
            unsigned long long ga =
                (unsigned long long)(size_t)(aSrc + (size_t)row * Kd + kc);
            asm volatile("global_load_async_to_lds_b128 %0, %1, off"
                         :: "v"(la), "v"(ga) : "memory");
            unsigned lw = (unsigned)(size_t)&sW[buf][row][kc];
            unsigned long long gw =
                (unsigned long long)(size_t)(wSrc + (size_t)row * Kd + kc);
            asm volatile("global_load_async_to_lds_b128 %0, %1, off"
                         :: "v"(lw), "v"(gw) : "memory");
        }
    };
    // ISA 16-bit A layout: lane r(<16) row m0+r, K = {half*8+0..7, 16+half*8+0..7}
    auto fragA = [&](int buf, int m0) -> v16bf {
        union { uint4 u[2]; v16bf v; } f;
        const bf16* p = &sA[buf][m0 + c16][0];
        f.u[0] = *reinterpret_cast<const uint4*>(p + half * 8);
        f.u[1] = *reinterpret_cast<const uint4*>(p + 16 + half * 8);
        return f.v;
    };
    // ISA 16-bit B layout: lane col = n0+c16, K = {half*8+0..7, 16+half*8+0..7}
    auto fragB = [&](int buf, int n0) -> v16bf {
        union { uint4 u[2]; v16bf v; } f;
        const bf16* p = &sW[buf][n0 + c16][0];
        f.u[0] = *reinterpret_cast<const uint4*>(p + half * 8);
        f.u[1] = *reinterpret_cast<const uint4*>(p + 16 + half * 8);
        return f.v;
    };

    v8f acc[4][2] = {};
    const int KT = Kd >> 5;

    stage(0, 0);
    asm volatile("s_wait_asynccnt 0x0" ::: "memory");
    __syncthreads();
    int buf = 0;
    for (int kk = 0; kk < KT; ++kk) {
        if (kk + 1 < KT) stage(buf ^ 1, kk + 1);
        // Pull tiles two K-steps ahead toward L2 (global_prefetch_b8).
        if (kk + 2 < KT) {
            __builtin_prefetch(Ag + (size_t)(tileM * 128 + (tid >> 1)) * Kd
                                  + (size_t)(kk + 2) * 32, 0, 3);
            __builtin_prefetch(Wg + (size_t)(tileN * 128 + (tid >> 1)) * Kd
                                  + (size_t)(kk + 2) * 32, 0, 3);
        }
        // Preload ALL fragments, then 8 back-to-back WMMAs on the matrix pipe.
        v16bf bfr0 = fragB(buf, wn * 32);
        v16bf bfr1 = fragB(buf, wn * 32 + 16);
        v16bf afr[4];
#pragma unroll
        for (int mi = 0; mi < 4; ++mi) afr[mi] = fragA(buf, wm * 64 + mi * 16);
#pragma unroll
        for (int mi = 0; mi < 4; ++mi) {
            acc[mi][0] = __builtin_amdgcn_wmma_f32_16x16x32_bf16(
                false, afr[mi], false, bfr0, (short)0, acc[mi][0], false, false);
            acc[mi][1] = __builtin_amdgcn_wmma_f32_16x16x32_bf16(
                false, afr[mi], false, bfr1, (short)0, acc[mi][1], false, false);
        }
        // Own async writes must land before signaling the barrier.
        asm volatile("s_wait_asynccnt 0x0" ::: "memory");
        __syncthreads();
        buf ^= 1;
    }

    // Epilogue. C/D layout: VGPR i -> row = half*8 + i, col = c16 (within 16x16).
    if (epi == 0) {
#pragma unroll
        for (int nj = 0; nj < 2; ++nj) {
            int col = tileN * 128 + wn * 32 + nj * 16 + c16;
            float bb = bg[col];
#pragma unroll
            for (int mi = 0; mi < 4; ++mi) {
#pragma unroll
                for (int i = 0; i < 8; ++i) {
                    int row = tileM * 128 + wm * 64 + mi * 16 + half * 8 + i;
                    float v = acc[mi][nj][i] + bb;
                    v = fmaxf(v, 0.0f);
                    Cg[(size_t)row * N + col] = (bf16)v;
                }
            }
        }
    } else {
        // Encoder last layer: column partial sums of relu(h) for mean-pool.
        // Each lane owns a unique (g, tileM, slot, col) -> fully deterministic.
        int slot = wm * 2 + half;
#pragma unroll
        for (int nj = 0; nj < 2; ++nj) {
            int col = tileN * 128 + wn * 32 + nj * 16 + c16;
            float bb = bg[col];
            float ps = 0.0f;
#pragma unroll
            for (int mi = 0; mi < 4; ++mi)
#pragma unroll
                for (int i = 0; i < 8; ++i)
                    ps += fmaxf(acc[mi][nj][i] + bb, 0.0f);
            pp[(((size_t)g * tilesM + tileM) * 4 + slot) * N + col] = ps;
        }
    }
}

// ---------------------------------------------------------------------------
// Weight convert f32 -> bf16, TRANSPOSED with K zero-padding.
// src [batch][sr][N] -> dst [batch][N][Kp]   (dst[n][k] = src[k][n], k<sr)
// ---------------------------------------------------------------------------
__global__ void cvt_pad_T(const float* __restrict__ src, bf16* __restrict__ dst,
                          int batch, int sr, int N, int Kp)
{
    size_t idx = (size_t)blockIdx.x * blockDim.x + threadIdx.x;
    size_t total = (size_t)batch * N * Kp;
    if (idx >= total) return;
    int k = (int)(idx % Kp);
    size_t t = idx / Kp;
    int n = (int)(t % N);
    int b = (int)(t / N);
    float v = (k < sr) ? src[((size_t)b * sr + k) * N + n] : 0.0f;
    dst[idx] = (bf16)v;
}

// Encoder layer-0 input: [16][512][32] = [which*8+b][point][pad32(x|y)]
__global__ void build_encA0(const float* __restrict__ x_c, const float* __restrict__ y_c,
                            const float* __restrict__ x_t, const float* __restrict__ y_t,
                            bf16* __restrict__ A0)
{
    int idx = blockIdx.x * blockDim.x + threadIdx.x;
    if (idx >= 16 * NPTS) return;
    int n = idx % NPTS, gA = idx / NPTS;     // gA = which*8 + b
    int which = gA >> 3, b = gA & 7;
    const float* x = which ? x_t : x_c;
    const float* y = which ? y_t : y_c;
    bf16* d = A0 + (size_t)idx * 32;
    d[0] = (bf16)x[((size_t)b * NPTS + n) * 2 + 0];
    d[1] = (bf16)x[((size_t)b * NPTS + n) * 2 + 1];
    d[2] = (bf16)y[((size_t)b * NPTS + n) * 3 + 0];
    d[3] = (bf16)y[((size_t)b * NPTS + n) * 3 + 1];
    d[4] = (bf16)y[((size_t)b * NPTS + n) * 3 + 2];
#pragma unroll
    for (int c = 5; c < 32; ++c) d[c] = (bf16)0.0f;
}

// Reduce per-tile pool partials: pp[128][4][4][512] -> pool[128][512] (raw sums)
__global__ void pool_reduce(const float* __restrict__ pp, float* __restrict__ pool)
{
    int idx = blockIdx.x * blockDim.x + threadIdx.x;
    if (idx >= 128 * Hsz) return;
    int g = idx >> 9, col = idx & 511;
    float s = 0.0f;
#pragma unroll
    for (int t = 0; t < 16; ++t)
        s += pp[(((size_t)g * 4 + (t >> 2)) * 4 + (t & 3)) * Hsz + col];
    pool[idx] = s;
}

// mu/logvar heads: r = pool/512;  mu = r@eWmu[k]+ebmu[k], lv likewise -> d_out
__global__ void mu_lv_head(const float* __restrict__ pool,
                           const float* __restrict__ eWmu, const float* __restrict__ ebmu,
                           const float* __restrict__ eWlv, const float* __restrict__ eblv,
                           float* __restrict__ out)
{
    int idx = blockIdx.x * blockDim.x + threadIdx.x;
    if (idx >= 128 * Lsz) return;
    int l = idx & 127, g = idx >> 7;          // g = which*64 + b*8 + k
    int k = g & 7, which = g >> 6;
    const float* r  = pool + (size_t)g * Hsz;
    const float* wm = eWmu + (size_t)k * Hsz * Lsz + l;
    const float* wl = eWlv + (size_t)k * Hsz * Lsz + l;
    float mu = 0.0f, lv = 0.0f;
    for (int h = 0; h < Hsz; ++h) {
        float rv = r[h] * (1.0f / (float)NPTS);
        mu = fmaf(rv, wm[(size_t)h * Lsz], mu);
        lv = fmaf(rv, wl[(size_t)h * Lsz], lv);
    }
    mu += ebmu[k * Lsz + l];
    lv += eblv[k * Lsz + l];
    size_t o = (size_t)(g & 63) * Lsz + l;
    out[(which ? OUT_MU_T : OUT_MU_C) + o] = mu;
    out[(which ? OUT_LV_T : OUT_LV_C) + o] = lv;
}

// z = mu_t + eps * exp(0.5*lv_t)
__global__ void z_kernel(const float* __restrict__ out, const float* __restrict__ eps,
                         float* __restrict__ z)
{
    int idx = blockIdx.x * blockDim.x + threadIdx.x;
    if (idx >= Bsz * Ksz * Lsz) return;
    z[idx] = out[OUT_MU_T + idx] + eps[idx] * expf(0.5f * out[OUT_LV_T + idx]);
}

// Build gate/decoder layer-0 inputs: xyz [65536][160], xz [65536][160]
__global__ void build_gateA0(const float* __restrict__ x_pred, const float* __restrict__ y_pred,
                             const float* __restrict__ z,
                             bf16* __restrict__ xyz, bf16* __restrict__ xz)
{
    int r = blockIdx.x * blockDim.x + threadIdx.x;
    if (r >= ROWS) return;
    int k = r & 7, bt = r >> 3;               // bt = b*T + t
    const float* xp = x_pred + (size_t)bt * 2;
    const float* yp = y_pred + (size_t)bt * 3;
    const float* zv = z + (size_t)(((bt >> 10) << 3) + k) * Lsz;  // b*8+k
    bf16* d1 = xyz + (size_t)r * 160;
    bf16* d2 = xz  + (size_t)r * 160;
    d1[0] = (bf16)xp[0]; d1[1] = (bf16)xp[1];
    d1[2] = (bf16)yp[0]; d1[3] = (bf16)yp[1]; d1[4] = (bf16)yp[2];
    d2[0] = (bf16)xp[0]; d2[1] = (bf16)xp[1];
#pragma unroll 8
    for (int l = 0; l < Lsz; ++l) {
        bf16 zl = (bf16)zv[l];
        d1[5 + l] = zl;
        d2[2 + l] = zl;
    }
#pragma unroll
    for (int c = 133; c < 160; ++c) d1[c] = (bf16)0.0f;
#pragma unroll
    for (int c = 130; c < 160; ++c) d2[c] = (bf16)0.0f;
}

// Gate output head (N=1 GEMV) — logits[r] = h[r]@Wo + bo
__global__ void gate_logits(const bf16* __restrict__ h, const float* __restrict__ Wo,
                            const float* __restrict__ bo, float* __restrict__ logits)
{
    int r = blockIdx.x * blockDim.x + threadIdx.x;
    if (r >= ROWS) return;
    const bf16* p = h + (size_t)r * HGsz;
    float s = bo[0];
    for (int c = 0; c < HGsz; ++c) s = fmaf((float)p[c], Wo[c], s);
    logits[r] = s;
}

// softmax over K=8 -> alpha region of d_out
__global__ void softmax_k(const float* __restrict__ logits, float* __restrict__ out, int off)
{
    int bt = blockIdx.x * blockDim.x + threadIdx.x;
    if (bt >= Bsz * Tsz) return;
    const float* lg = logits + (size_t)bt * Ksz;
    float m = lg[0];
#pragma unroll
    for (int k = 1; k < Ksz; ++k) m = fmaxf(m, lg[k]);
    float e[Ksz], s = 0.0f;
#pragma unroll
    for (int k = 0; k < Ksz; ++k) { e[k] = expf(lg[k] - m); s += e[k]; }
    float inv = 1.0f / s;
#pragma unroll
    for (int k = 0; k < Ksz; ++k) out[off + (size_t)bt * Ksz + k] = e[k] * inv;
}

// Decoder output head: 512->6, fused sigmoid/softplus -> d_out
__global__ void dec_out(const bf16* __restrict__ h, const float* __restrict__ dWo,
                        const float* __restrict__ dbo, float* __restrict__ out)
{
    int r = blockIdx.x * blockDim.x + threadIdx.x;
    if (r >= ROWS) return;
    const bf16* p = h + (size_t)r * DHsz;
    float o[6];
#pragma unroll
    for (int j = 0; j < 6; ++j) o[j] = dbo[j];
    for (int c = 0; c < DHsz; ++c) {
        float hv = (float)p[c];
        const float* w = dWo + (size_t)c * 6;
#pragma unroll
        for (int j = 0; j < 6; ++j) o[j] = fmaf(hv, w[j], o[j]);
    }
#pragma unroll
    for (int j = 0; j < 3; ++j)
        out[OUT_YMEAN + (size_t)r * 3 + j] = 1.0f / (1.0f + expf(-o[j]));
#pragma unroll
    for (int j = 0; j < 3; ++j) {
        float x = o[3 + j];
        float sp = (x > 20.0f) ? x : log1pf(expf(x));
        out[OUT_YSTD + (size_t)r * 3 + j] = sp;
    }
}

// ---------------------------------------------------------------------------
// Host launcher
// ---------------------------------------------------------------------------
static inline int gblk(long long n) { return (int)((n + 255) / 256); }

static void launch_gemm(const bf16* A, const bf16* WT, const float* bias, bf16* C,
                        float* pp, int M, int Kd, int N, int G, int aShift, int wMask,
                        int epi, hipStream_t s)
{
    int tilesM = M / 128, tilesN = N / 128;
    dim3 grid((unsigned)(tilesM * tilesN), (unsigned)G);
    gemm_bf16<<<grid, 256, 0, s>>>(A, WT, bias, C, pp, M, Kd, N, tilesM, aShift, wMask, epi);
}

extern "C" void kernel_launch(void* const* d_in, const int* in_sizes, int n_in,
                              void* d_out, int out_size, void* d_ws, size_t ws_size,
                              hipStream_t stream)
{
    (void)in_sizes; (void)n_in; (void)out_size; (void)ws_size;

    const float* x_c    = (const float*)d_in[0];
    const float* y_c    = (const float*)d_in[1];
    const float* x_t    = (const float*)d_in[2];
    const float* y_t    = (const float*)d_in[3];
    const float* x_pred = (const float*)d_in[4];
    const float* y_pred = (const float*)d_in[5];
    const float* eps    = (const float*)d_in[6];
    const float* eW0    = (const float*)d_in[9];
    const float* eb0    = (const float*)d_in[10];
    const float* eWh    = (const float*)d_in[11];
    const float* ebh    = (const float*)d_in[12];
    const float* eWmu   = (const float*)d_in[13];
    const float* ebmu   = (const float*)d_in[14];
    const float* eWlv   = (const float*)d_in[15];
    const float* eblv   = (const float*)d_in[16];
    const float* pW0    = (const float*)d_in[17];
    const float* pb0    = (const float*)d_in[18];
    const float* pWh    = (const float*)d_in[19];
    const float* pbh    = (const float*)d_in[20];
    const float* pWo    = (const float*)d_in[21];
    const float* pbo    = (const float*)d_in[22];
    const float* qW0    = (const float*)d_in[23];
    const float* qb0    = (const float*)d_in[24];
    const float* qWh    = (const float*)d_in[25];
    const float* qbh    = (const float*)d_in[26];
    const float* qWo    = (const float*)d_in[27];
    const float* qbo    = (const float*)d_in[28];
    const float* dW0    = (const float*)d_in[29];
    const float* db0    = (const float*)d_in[30];
    const float* dWh    = (const float*)d_in[31];
    const float* dbh    = (const float*)d_in[32];
    const float* dWo    = (const float*)d_in[33];
    const float* dbo    = (const float*)d_in[34];
    float* out = (float*)d_out;

    // Workspace carve-up (~193 MB total)
    char* ws = (char*)d_ws;
    size_t o = 0;
    auto alloc = [&](size_t bytes) { size_t r = o; o = (o + bytes + 255) & ~(size_t)255; return r; };
    bf16*  wEW0 = (bf16*)(ws + alloc((size_t)8 * 512 * 32 * 2));      // [K][N=512][Kp=32]
    bf16*  wEWH = (bf16*)(ws + alloc((size_t)16 * 512 * 512 * 2));    // [2*K][512][512]
    bf16*  wPW0 = (bf16*)(ws + alloc((size_t)256 * 160 * 2));
    bf16*  wPWH = (bf16*)(ws + alloc((size_t)256 * 256 * 2));
    bf16*  wQW0 = (bf16*)(ws + alloc((size_t)256 * 160 * 2));
    bf16*  wQWH = (bf16*)(ws + alloc((size_t)256 * 256 * 2));
    bf16*  wDW0 = (bf16*)(ws + alloc((size_t)512 * 160 * 2));
    bf16*  wDWH = (bf16*)(ws + alloc((size_t)3 * 512 * 512 * 2));
    float* pp   = (float*)(ws + alloc((size_t)128 * 16 * 512 * 4));
    float* pool = (float*)(ws + alloc((size_t)128 * 512 * 4));
    float* zbuf = (float*)(ws + alloc((size_t)8192 * 4));
    float* logP = (float*)(ws + alloc((size_t)ROWS * 4));
    float* logQ = (float*)(ws + alloc((size_t)ROWS * 4));
    bf16*  encA0= (bf16*)(ws + alloc((size_t)16 * 512 * 32 * 2));
    bf16*  xyzA = (bf16*)(ws + alloc((size_t)ROWS * 160 * 2));
    bf16*  xzA  = (bf16*)(ws + alloc((size_t)ROWS * 160 * 2));
    bf16*  act0 = (bf16*)(ws + alloc((size_t)ROWS * 512 * 2));  // also encoder acts (67MB)
    bf16*  act1 = (bf16*)(ws + alloc((size_t)ROWS * 512 * 2));

    // 1) weight conversion: f32 -> bf16, transposed to [N][Kpad] per matrix
    cvt_pad_T<<<gblk((long long)8*512*32),   256, 0, stream>>>(eW0, wEW0, 8, 5, 512, 32);
    cvt_pad_T<<<gblk((long long)16*512*512), 256, 0, stream>>>(eWh, wEWH, 16, 512, 512, 512);
    cvt_pad_T<<<gblk((long long)256*160),    256, 0, stream>>>(pW0, wPW0, 1, 133, 256, 160);
    cvt_pad_T<<<gblk((long long)256*256),    256, 0, stream>>>(pWh, wPWH, 1, 256, 256, 256);
    cvt_pad_T<<<gblk((long long)256*160),    256, 0, stream>>>(qW0, wQW0, 1, 130, 256, 160);
    cvt_pad_T<<<gblk((long long)256*256),    256, 0, stream>>>(qWh, wQWH, 1, 256, 256, 256);
    cvt_pad_T<<<gblk((long long)512*160),    256, 0, stream>>>(dW0, wDW0, 1, 130, 512, 160);
    cvt_pad_T<<<gblk((long long)3*512*512),  256, 0, stream>>>(dWh, wDWH, 3, 512, 512, 512);

    // 2) Encoder: grouped over g = which*64 + b*8 + k (G=128, A shared across k)
    build_encA0<<<gblk(16 * NPTS), 256, 0, stream>>>(x_c, y_c, x_t, y_t, encA0);
    launch_gemm(encA0, wEW0,               eb0,        act0, pp, NPTS, 32,  Hsz, 128, 3, 7, 0, stream);
    launch_gemm(act0,  wEWH,               ebh,        act1, pp, NPTS, Hsz, Hsz, 128, 0, 7, 0, stream);
    launch_gemm(act1,  wEWH + (size_t)8*Hsz*Hsz, ebh + 8*Hsz, act0, pp, NPTS, Hsz, Hsz, 128, 0, 7, 1, stream);
    pool_reduce<<<gblk(128 * Hsz), 256, 0, stream>>>(pp, pool);
    mu_lv_head<<<gblk(128 * Lsz), 256, 0, stream>>>(pool, eWmu, ebmu, eWlv, eblv, out);
    z_kernel<<<gblk(Bsz * Ksz * Lsz), 256, 0, stream>>>(out, eps, zbuf);

    // 3) Gate/decoder layer-0 inputs (xz shared by prior gate and decoder)
    build_gateA0<<<gblk(ROWS), 256, 0, stream>>>(x_pred, y_pred, zbuf, xyzA, xzA);

    // 4) Posterior gate
    launch_gemm(xyzA, wPW0, pb0, act0, pp, ROWS, 160,  HGsz, 1, 0, 0, 0, stream);
    launch_gemm(act0, wPWH, pbh, act1, pp, ROWS, HGsz, HGsz, 1, 0, 0, 0, stream);
    gate_logits<<<gblk(ROWS), 256, 0, stream>>>(act1, pWo, pbo, logP);
    softmax_k<<<gblk(Bsz * Tsz), 256, 0, stream>>>(logP, out, OUT_APOST);

    // 5) Prior gate
    launch_gemm(xzA,  wQW0, qb0, act0, pp, ROWS, 160,  HGsz, 1, 0, 0, 0, stream);
    launch_gemm(act0, wQWH, qbh, act1, pp, ROWS, HGsz, HGsz, 1, 0, 0, 0, stream);
    gate_logits<<<gblk(ROWS), 256, 0, stream>>>(act1, qWo, qbo, logQ);
    softmax_k<<<gblk(Bsz * Tsz), 256, 0, stream>>>(logQ, out, OUT_APRIOR);

    // 6) Decoder (the FLOP heavyweight: 4 x [65536 x 512 x 512]-class GEMMs)
    launch_gemm(xzA,  wDW0,                 db0,          act0, pp, ROWS, 160,  DHsz, 1, 0, 0, 0, stream);
    launch_gemm(act0, wDWH,                 dbh,          act1, pp, ROWS, DHsz, DHsz, 1, 0, 0, 0, stream);
    launch_gemm(act1, wDWH + (size_t)DHsz*DHsz,   dbh + DHsz,   act0, pp, ROWS, DHsz, DHsz, 1, 0, 0, 0, stream);
    launch_gemm(act0, wDWH + (size_t)2*DHsz*DHsz, dbh + 2*DHsz, act1, pp, ROWS, DHsz, DHsz, 1, 0, 0, 0, stream);
    dec_out<<<gblk(ROWS), 256, 0, stream>>>(act1, dWo, dbo, out);
}
